// RNN_25529285607659
// MI455X (gfx1250) — compile-verified, matrix operand order checked by reading, not run on hardware
//
#include <hip/hip_runtime.h>

typedef __bf16 bf16_t;
typedef __attribute__((ext_vector_type(16))) __bf16 v16bf;
typedef __attribute__((ext_vector_type(8)))  __bf16 v8bf;
typedef __attribute__((ext_vector_type(8)))  float  v8f;

#define LDSS 40  // A/B-staging LDS row stride (bf16): 32 data + 8 pad

__device__ __forceinline__ float fast_tanh(float x) {
  // branchless; saturates to +/-1 for large |x| (exp -> inf or 0)
  const float e = __expf(2.0f * x);
  return 1.0f - 2.0f / (e + 1.0f);
}

// Async global->LDS copy of one 32B chunk per lane (2 x B128), ASYNCcnt-tracked.
// LDS aperture keeps the byte offset in addr[31:0]; inst offset applies to both
// the global and LDS addresses, so one (lds, global) pair covers both halves.
__device__ __forceinline__ void async_copy32(const bf16_t* g, bf16_t* l) {
  const unsigned la = (unsigned)(uintptr_t)l;
  asm volatile(
      "global_load_async_to_lds_b128 %0, %1, off\n\t"
      "global_load_async_to_lds_b128 %0, %1, off offset:16"
      :: "v"(la), "v"(g) : "memory");
}
__device__ __forceinline__ void wait_async0() {
  asm volatile("s_wait_asynccnt 0" ::: "memory");
}

// ---------------------------------------------------------------------------
// bf16 WMMA GEMM: C[M,NN] = A[M,K] * Bm[NN,K]^T (+bias) -> f32/bf16
// Block tile 128x128, 8 waves of 64x32. Double-buffered LDS staged with
// GLOBAL_LOAD_ASYNC_TO_LDS; 1 barrier + 1 asynccnt wait per 32-wide K-slab.
// ---------------------------------------------------------------------------
template <int NN, bool HAS_BIAS, bool HAS_O32, bool HAS_O16>
__global__ __launch_bounds__(256)
void wmma_gemm_kernel(const bf16_t* __restrict__ A,
                      const bf16_t* __restrict__ Bm,
                      const float*  __restrict__ bias,
                      float*        __restrict__ out32,
                      bf16_t*       __restrict__ out16,
                      int K)
{
  __shared__ bf16_t As[2][128 * LDSS];
  __shared__ bf16_t Bs[2][128 * LDSS];

  const int tid  = threadIdx.x;
  const int lane = tid & 31;
  const int wave = tid >> 5;
  const int wm   = wave >> 2;            // 0..1 : 64-row slab
  const int wn   = wave & 3;             // 0..3 : 32-col slab
  const int tileM = blockIdx.y * 128;
  const int tileN = blockIdx.x * 128;

  const int r     = lane & 15;
  const int khalf = (lane >> 4) * 8;

  const v8f vzero = {0.f,0.f,0.f,0.f,0.f,0.f,0.f,0.f};
  v8f acc[4][2];
  #pragma unroll
  for (int mt = 0; mt < 4; ++mt)
    #pragma unroll
    for (int nt = 0; nt < 2; ++nt)
      acc[mt][nt] = vzero;

  const int rowL = tid >> 1;             // 0..127
  const int ch   = (tid & 1) * 16;       // 0 or 16 bf16 = 32B
  const bf16_t* gA = A  + (size_t)(tileM + rowL) * K + ch;
  const bf16_t* gB = Bm + (size_t)(tileN + rowL) * K + ch;

  auto stage = [&](int kk, int b) {
    async_copy32(gA + kk, As[b] + rowL * LDSS + ch);
    async_copy32(gB + kk, Bs[b] + rowL * LDSS + ch);
  };

  const int nsteps = K >> 5;
  stage(0, 0);

  int buf = 0;
  for (int i = 0; i < nsteps; ++i) {
    wait_async0();                       // slab i landed in LDS (this wave)
    __syncthreads();                     // all waves landed; prior reads done
    if (i + 1 < nsteps) stage((i + 1) << 5, buf ^ 1); // overlaps compute below

    const bf16_t* Ab = As[buf];
    const bf16_t* Bb = Bs[buf];
    v16bf af[4], bfv[2];
    #pragma unroll
    for (int mt = 0; mt < 4; ++mt) {
      const bf16_t* p = Ab + (wm * 64 + mt * 16 + r) * LDSS;
      v8bf lo = *(const v8bf*)(p + khalf);
      v8bf hi = *(const v8bf*)(p + 16 + khalf);
      af[mt] = __builtin_shufflevector(lo, hi, 0,1,2,3,4,5,6,7,8,9,10,11,12,13,14,15);
    }
    #pragma unroll
    for (int nt = 0; nt < 2; ++nt) {
      const bf16_t* p = Bb + (wn * 32 + nt * 16 + r) * LDSS;
      v8bf lo = *(const v8bf*)(p + khalf);
      v8bf hi = *(const v8bf*)(p + 16 + khalf);
      bfv[nt] = __builtin_shufflevector(lo, hi, 0,1,2,3,4,5,6,7,8,9,10,11,12,13,14,15);
    }
    #pragma unroll
    for (int mt = 0; mt < 4; ++mt)
      #pragma unroll
      for (int nt = 0; nt < 2; ++nt)
        acc[mt][nt] = __builtin_amdgcn_wmma_f32_16x16x32_bf16(
            false, af[mt], false, bfv[nt], (short)0, acc[mt][nt], false, false);

    buf ^= 1;
  }

  const int rbase = tileM + wm * 64 + ((lane >> 4) << 3);
  const int cbase = tileN + wn * 32 + (lane & 15);
  #pragma unroll
  for (int nt = 0; nt < 2; ++nt) {
    const int c = cbase + nt * 16;
    const size_t base = (size_t)rbase * NN + c;
    float bv = 0.0f;
    if constexpr (HAS_BIAS) bv = bias[c];
    float*  o32 = nullptr;
    bf16_t* o16 = nullptr;
    if constexpr (HAS_O32) o32 = out32 + base;
    if constexpr (HAS_O16) o16 = out16 + base;
    #pragma unroll
    for (int mt = 0; mt < 4; ++mt) {
      #pragma unroll
      for (int j = 0; j < 8; ++j) {
        const int ofs = (mt * 16 + j) * NN;
        float v = acc[mt][nt][j] + bv;
        if constexpr (HAS_O32) o32[ofs] = v;
        if constexpr (HAS_O16) o16[ofs] = (bf16_t)v;
      }
    }
  }
}

// ---------------------------------------------------------------------------
// Persistent recurrence kernel: h_{t+1} = tanh(pre_t + h_t @ Whh^T), t = 0..T-1
// Grid must be (8,2) = 16 blocks, 256 threads. Each block keeps its 128-row
// slice of Whh (N-major) RESIDENT in LDS (~258KB of the 320KB WGP LDS) and
// iterates all timesteps internally; blocks sync via device-scope atomics.
// h slabs are staged with async global->LDS copies.
// ---------------------------------------------------------------------------
#define BSTR 1032  // resident-B LDS row stride (1024 data + 8 pad), 2064B rows
#define SCAN_LDS_BYTES (128 * BSTR * 2 + 2 * 128 * LDSS * 2)

__global__ __launch_bounds__(256)
void scan_persistent_kernel(const bf16_t* __restrict__ Whh,  // (H,H) N-major
                            const float*  __restrict__ pre,  // (T,B,H)
                            float*        __restrict__ hidden, // (T,B,H)
                            bf16_t*       __restrict__ hb0,  // (B,H) ping
                            bf16_t*       __restrict__ hb1,  // (B,H) pong
                            unsigned*     __restrict__ ctr,  // zeroed counter
                            int T)
{
  constexpr int Hc = 1024;
  extern __shared__ char smem[];
  bf16_t* Bsl = (bf16_t*)smem;                               // 128 x BSTR resident
  bf16_t* Ast = (bf16_t*)(smem + (size_t)128 * BSTR * 2);    // 2 x 128 x LDSS staging

  const int tid  = threadIdx.x;
  const int lane = tid & 31;
  const int wave = tid >> 5;
  const int wm   = wave >> 2;
  const int wn   = wave & 3;
  const int tileM = blockIdx.y * 128;
  const int tileN = blockIdx.x * 128;
  const int r     = lane & 15;
  const int khalf = (lane >> 4) * 8;
  const int nblocks = gridDim.x * gridDim.y;

  // ---- preload resident Whh slice: rows [tileN, tileN+128) x K=1024 ----
  for (int u = tid; u < 128 * (Hc / 16); u += 256) {
    const int row = u >> 6, chunk = (u & 63) * 16;
    const uint4* s = (const uint4*)(Whh + (size_t)(tileN + row) * Hc + chunk);
    uint4* d = (uint4*)(Bsl + row * BSTR + chunk);
    d[0] = s[0]; d[1] = s[1];
  }
  __syncthreads();

  const int rowL = tid >> 1;
  const int ch   = (tid & 1) * 16;
  const v8f vzero = {0.f,0.f,0.f,0.f,0.f,0.f,0.f,0.f};

  const int rbase = tileM + wm * 64 + ((lane >> 4) << 3);
  const int cbase = tileN + wn * 32 + (lane & 15);

  for (int t = 0; t < T; ++t) {
    const bf16_t* hA = ((t & 1) ? hb1 : hb0) + (size_t)(tileM + rowL) * Hc + ch;
    bf16_t*      hOut = (t & 1) ? hb0 : hb1;
    const float* preT = pre    + (size_t)t * 256 * Hc;
    float*       hidT = hidden + (size_t)t * 256 * Hc;

    v8f acc[4][2];
    #pragma unroll
    for (int mt = 0; mt < 4; ++mt)
      #pragma unroll
      for (int nt = 0; nt < 2; ++nt)
        acc[mt][nt] = vzero;

    auto stageA = [&](int kk, int b) {
      async_copy32(hA + kk, Ast + b * (128 * LDSS) + rowL * LDSS + ch);
    };

    stageA(0, 0);   // safe: Ast[0] last read before previous grid sync

    int buf = 0;
    for (int i = 0; i < Hc / 32; ++i) {
      wait_async0();
      __syncthreads();
      if (i + 1 < Hc / 32) stageA((i + 1) << 5, buf ^ 1);

      const bf16_t* Ab = Ast + buf * (128 * LDSS);
      const int k0 = i * 32;
      v16bf af[4], bfv[2];
      #pragma unroll
      for (int mt = 0; mt < 4; ++mt) {
        const bf16_t* p = Ab + (wm * 64 + mt * 16 + r) * LDSS;
        v8bf lo = *(const v8bf*)(p + khalf);
        v8bf hi = *(const v8bf*)(p + 16 + khalf);
        af[mt] = __builtin_shufflevector(lo, hi, 0,1,2,3,4,5,6,7,8,9,10,11,12,13,14,15);
      }
      #pragma unroll
      for (int nt = 0; nt < 2; ++nt) {
        const bf16_t* p = Bsl + (wn * 32 + nt * 16 + r) * BSTR + k0;
        v8bf lo = *(const v8bf*)(p + khalf);
        v8bf hi = *(const v8bf*)(p + 16 + khalf);
        bfv[nt] = __builtin_shufflevector(lo, hi, 0,1,2,3,4,5,6,7,8,9,10,11,12,13,14,15);
      }
      #pragma unroll
      for (int mt = 0; mt < 4; ++mt)
        #pragma unroll
        for (int nt = 0; nt < 2; ++nt)
          acc[mt][nt] = __builtin_amdgcn_wmma_f32_16x16x32_bf16(
              false, af[mt], false, bfv[nt], (short)0, acc[mt][nt], false, false);

      buf ^= 1;
    }

    // epilogue: tanh(pre + acc) -> hidden (f32) and h ping-pong (bf16)
    #pragma unroll
    for (int nt = 0; nt < 2; ++nt) {
      const int c = cbase + nt * 16;
      const size_t base = (size_t)rbase * Hc + c;
      const float* ap  = preT + base;
      float*       o32 = hidT + base;
      bf16_t*      o16 = hOut + base;
      #pragma unroll
      for (int mt = 0; mt < 4; ++mt) {
        #pragma unroll
        for (int j = 0; j < 8; ++j) {
          const int ofs = (mt * 16 + j) * Hc;
          float v = fast_tanh(acc[mt][nt][j] + ap[ofs]);
          o32[ofs] = v;
          o16[ofs] = (bf16_t)v;
        }
      }
    }

    // ---- grid-wide sync: release our h stores, wait for all 16 blocks ----
    __threadfence();                 // per-thread: stores visible at device scope
    __syncthreads();
    if (tid == 0) {
      __hip_atomic_fetch_add(ctr, 1u, __ATOMIC_ACQ_REL, __HIP_MEMORY_SCOPE_AGENT);
      const unsigned target = (unsigned)nblocks * (unsigned)(t + 1);
      while (__hip_atomic_load(ctr, __ATOMIC_ACQUIRE, __HIP_MEMORY_SCOPE_AGENT) < target)
        __builtin_amdgcn_s_sleep(2);
    }
    __syncthreads();
  }
}

// ---------------------------------------------------------------------------
// x (B,I,T) fp32 -> xt (T*B, I) bf16, LDS-tiled transpose over the (i,t) plane
// ---------------------------------------------------------------------------
__global__ __launch_bounds__(256)
void transpose_x_kernel(const float* __restrict__ x, bf16_t* __restrict__ xt,
                        int B, int I, int T)
{
  __shared__ float tile[32][33];
  const int t0 = blockIdx.x * 32;
  const int i0 = blockIdx.y * 32;
  const int b  = blockIdx.z;
  const int tx = threadIdx.x & 31;
  const int ty = threadIdx.x >> 5;
  const float* xb = x + (size_t)b * I * T;
  #pragma unroll
  for (int s = 0; s < 4; ++s)
    tile[ty + s * 8][tx] = xb[(size_t)(i0 + ty + s * 8) * T + t0 + tx];
  __syncthreads();
  #pragma unroll
  for (int s = 0; s < 4; ++s) {
    const int tt = ty + s * 8;
    xt[((size_t)(t0 + tt) * B + b) * I + i0 + tx] = (bf16_t)tile[tx][tt];
  }
}

// ---------------------------------------------------------------------------
__global__ __launch_bounds__(256)
void convert_w_kernel(const float* __restrict__ src, bf16_t* __restrict__ dst,
                      int rows, int cols, int ld, int off)
{
  const int idx = blockIdx.x * 256 + threadIdx.x;
  if (idx >= rows * cols) return;
  const int rr = idx / cols, cc = idx % cols;
  dst[idx] = (bf16_t)src[(size_t)rr * ld + off + cc];
}

// ---------------------------------------------------------------------------
__global__ __launch_bounds__(256)
void bn_norm_kernel(const float* __restrict__ hidden,
                    const float* __restrict__ gamma,
                    const float* __restrict__ beta,
                    bf16_t* __restrict__ hn,
                    int T, int BH, int H)
{
  const int idx = blockIdx.x * 256 + threadIdx.x;
  if (idx >= BH) return;
  const int h = idx % H;
  float sum = 0.f, sq = 0.f;
  for (int t = 0; t < T; ++t) {
    const float v = hidden[(size_t)t * BH + idx];
    sum += v; sq += v * v;
  }
  const float mean = sum / T;
  const float var  = sq / T - mean * mean;
  const float rstd = rsqrtf(var + 1e-5f);
  const float g  = gamma[h] * rstd;
  const float bt = beta[h];
  for (int t = 0; t < T; ++t) {
    const float v = hidden[(size_t)t * BH + idx];
    hn[(size_t)t * BH + idx] = (bf16_t)((v - mean) * g + bt);
  }
}

// ---------------------------------------------------------------------------
extern "C" void kernel_launch(void* const* d_in, const int* in_sizes, int n_in,
                              void* d_out, int out_size, void* d_ws, size_t ws_size,
                              hipStream_t stream)
{
  (void)in_sizes; (void)n_in; (void)out_size; (void)ws_size;
  const int B = 256, I = 512, T = 256, H = 1024;
  const int TB = T * B;

  const float* x     = (const float*)d_in[0];
  const float* Wi    = (const float*)d_in[1];
  const float* bi    = (const float*)d_in[2];
  const float* Wh    = (const float*)d_in[3];
  const float* bh    = (const float*)d_in[4];
  const float* gamma = (const float*)d_in[5];
  const float* beta  = (const float*)d_in[6];
  const float* Wo    = (const float*)d_in[7];
  const float* bo    = (const float*)d_in[8];
  float* out = (float*)d_out;

  size_t off = 0;
  auto alloc = [&](size_t sz) -> void* {
    void* p = (char*)d_ws + off;
    off += (sz + 255) & ~(size_t)255;
    return p;
  };
  bf16_t*   Wi_bf   = (bf16_t*)alloc((size_t)H * I * 2);
  bf16_t*   Whin_bf = (bf16_t*)alloc((size_t)H * H * 2);
  bf16_t*   Whh_bf  = (bf16_t*)alloc((size_t)H * H * 2);
  bf16_t*   Wo_bf   = (bf16_t*)alloc((size_t)I * H * 2);
  float*    pre     = (float*)alloc((size_t)TB * H * 4);
  float*    hidden  = (float*)alloc((size_t)TB * H * 4);
  bf16_t*   hbuf0   = (bf16_t*)alloc((size_t)B * H * 2);
  bf16_t*   hbuf1   = (bf16_t*)alloc((size_t)B * H * 2);
  unsigned* ctr     = (unsigned*)alloc(256);

  // aliases: xt/inp live in the hidden region (only written from the scan on);
  // hn reuses the pre region (pre is dead after the scan).
  bf16_t* xt     = (bf16_t*)hidden;                               // TB*I bf16
  bf16_t* inp_bf = (bf16_t*)((char*)hidden + (size_t)TB * I * 2); // TB*H bf16
  bf16_t* hn     = (bf16_t*)pre;                                  // TB*H bf16

  // opt-in to >64KB dynamic LDS for the persistent scan kernel (idempotent)
  (void)hipFuncSetAttribute((const void*)scan_persistent_kernel,
                            hipFuncAttributeMaxDynamicSharedMemorySize,
                            SCAN_LDS_BYTES);

  // weights -> bf16 (Wh split: Wh_in = Wh[:, :H], Wh_hid = Wh[:, H:])
  convert_w_kernel<<<(H * I + 255) / 256, 256, 0, stream>>>(Wi, Wi_bf, H, I, I, 0);
  convert_w_kernel<<<(H * H + 255) / 256, 256, 0, stream>>>(Wh, Whin_bf, H, H, 2 * H, 0);
  convert_w_kernel<<<(H * H + 255) / 256, 256, 0, stream>>>(Wh, Whh_bf, H, H, 2 * H, H);
  convert_w_kernel<<<(I * H + 255) / 256, 256, 0, stream>>>(Wo, Wo_bf, I, H, H, 0);

  // x (B,I,T) -> xt (T*B, I) bf16
  transpose_x_kernel<<<dim3(T / 32, I / 32, B), 256, 0, stream>>>(x, xt, B, I, T);

  // inp = xt @ Wi^T + bi  -> bf16 (TB x H)
  wmma_gemm_kernel<1024, true, false, true>
      <<<dim3(H / 128, TB / 128), 256, 0, stream>>>(
      xt, Wi_bf, bi, nullptr, inp_bf, I);

  // pre = inp @ Wh_in^T + bh -> fp32 (TB x H)
  wmma_gemm_kernel<1024, true, true, false>
      <<<dim3(H / 128, TB / 128), 256, 0, stream>>>(
      inp_bf, Whin_bf, bh, pre, nullptr, H);

  // recurrence: single persistent kernel, 16 blocks, Whh resident in LDS
  hipMemsetAsync(hbuf0, 0, (size_t)B * H * 2, stream);
  hipMemsetAsync(ctr, 0, sizeof(unsigned), stream);
  scan_persistent_kernel<<<dim3(H / 128, B / 128), 256, SCAN_LDS_BYTES, stream>>>(
      Whh_bf, pre, hidden, hbuf0, hbuf1, ctr, T);

  // batchnorm over T (population var), normalized hidden -> bf16
  bn_norm_kernel<<<(B * H + 255) / 256, 256, 0, stream>>>(
      hidden, gamma, beta, hn, T, B * H, H);

  // out = hn @ Wo^T + bo -> fp32 (TB x I) == (T,B,I)
  wmma_gemm_kernel<512, true, true, false>
      <<<dim3(I / 128, TB / 128), 256, 0, stream>>>(
      hn, Wo_bf, bo, out, nullptr, H);
}
